// AttnBlock_21895743275587
// MI455X (gfx1250) — matrix operand order repaired and checked by reading, not compiled, not run on hardware
//
#include <hip/hip_runtime.h>
#include <hip/hip_bf16.h>

// ---- CDNA5 wave32 WMMA types ----
typedef __attribute__((ext_vector_type(16))) __bf16 v16bf;
typedef __attribute__((ext_vector_type(8)))  float  v8f;
typedef __attribute__((ext_vector_type(2)))  float  v2f;

#define BATCH 16
#define CH    256
#define NPIX  4096     // 64*64
#define INNER 128
#define HEADS 4
#define DH    32

// Workspace layout (bytes). Total ~151.6 MB.
#define OFF_NORM  ((size_t)0)            // swizzled bf16 normed  [B][8 kc][256 nt][512]  = 33,554,432
#define OFF_QKV   ((size_t)33554432)     // fp32 qkv [B][384][N]                          = 100,663,296
#define OFF_ATTN  ((size_t)134217728)    // swizzled bf16 attn   [B][4 kc][256 nt][512]   = 16,777,216
#define OFF_CTX   ((size_t)150994944)    // fp32 ctx [B][4][32][32]                       = 262,144
#define OFF_KMAX  ((size_t)151257088)    // fp32 [B*128]                                  = 8,192
#define OFF_KSUM  ((size_t)151265280)    // fp32 [B*128]                                  = 8,192
#define OFF_WQKV  ((size_t)151273472)    // bf16 Wqkv A-frag [24 ot][8 kc][512]           = 196,608
#define OFF_WOUT  ((size_t)151470080)    // bf16 Wout A-frag [16 ot][4 kc][512]           = 65,536

__device__ __forceinline__ unsigned short f2bf(float f) {
    union { float f; unsigned int u; } v; v.f = f;
    unsigned int u = v.u;
    unsigned int r = u + 0x7FFFu + ((u >> 16) & 1u);   // round-to-nearest-even
    return (unsigned short)(r >> 16);
}

// ---------------------------------------------------------------------------
// Weight -> bf16 A-fragment swizzle.  A-frag (16x32 bf16, this code's contiguous k order):
// lane = 16*half + m ; element j: k = half*16 + j
// Stored so GEMM reads A[(ot*kchunks+kc)*512 + lane*16 + j] contiguously.
// ---------------------------------------------------------------------------
__global__ void k_wswz(const float* __restrict__ W, unsigned short* __restrict__ out,
                       int otiles, int kchunks, int rowlen) {
    int idx = blockIdx.x * blockDim.x + threadIdx.x;
    int total = otiles * kchunks * 512;
    if (idx >= total) return;
    int j    = idx & 15;
    int lane = (idx >> 4) & 31;
    int t    = idx >> 9;
    int kc   = t % kchunks;
    int ot   = t / kchunks;
    int m    = lane & 15;
    int half = lane >> 4;
    int k    = half * 16 + j;
    int o    = ot * 16 + m;
    int c    = kc * 32 + k;
    out[idx] = f2bf(W[(size_t)o * rowlen + c]);
}

// ---------------------------------------------------------------------------
// Channel LayerNorm of x, emit bf16 in B-fragment layout:
// B-frag (32x16 bf16): lane = 16*(k/16) + n%16 ; element j: k = (lane/16)*16+j
// ---------------------------------------------------------------------------
__global__ void k_ln_swizzle(const float* __restrict__ x, unsigned short* __restrict__ nrm) {
    int gid = blockIdx.x * blockDim.x + threadIdx.x;    // 65536 pixels
    int b = gid >> 12;
    int n = gid & (NPIX - 1);
    const float* xb = x + (size_t)b * CH * NPIX + n;
    float s = 0.f, s2 = 0.f;
    for (int c = 0; c < CH; ++c) { float v = xb[(size_t)c * NPIX]; s += v; s2 += v * v; }
    float mu  = s * (1.f / CH);
    float var = s2 * (1.f / CH) - mu * mu;
    float rs  = __frsqrt_rn(var + 1e-5f);
    unsigned short* nb = nrm + (size_t)b * 8 * 256 * 512;
    int ntile = n >> 4, lanecol = n & 15;
    for (int c = 0; c < CH; ++c) {
        float v = (xb[(size_t)c * NPIX] - mu) * rs;
        int kc = c >> 5, kin = c & 31;
        int lane = ((kin >> 4) << 4) | lanecol;
        int j = kin & 15;
        nb[(((size_t)kc * 256 + ntile) * 512) + lane * 16 + j] = f2bf(v);
    }
}

// ---------------------------------------------------------------------------
// GEMM1: qkv[b,o,n] = sum_c Wqkv[o,c]*normed[b,c,n].  bf16 WMMA, fp32 acc.
// One wave -> 32x64 output tile (2 A frags x 4 B frags, 8 wmma / k-step),
// software-pipelined: k-step kc+1 fragments are fetched before kc's wmmas.
// ---------------------------------------------------------------------------
__global__ void k_gemm_qkv(const unsigned short* __restrict__ Wsw,
                           const unsigned short* __restrict__ nrm,
                           float* __restrict__ qkv) {
    int wid  = (blockIdx.x * blockDim.x + threadIdx.x) >> 5;   // 0..12287
    int lane = threadIdx.x & 31;
    int b  = wid / (12 * 64);
    int r  = wid % (12 * 64);
    int op = r >> 6;          // 0..11  pair of output row tiles
    int ng = r & 63;          // group of 4 n-tiles
    const v16bf* A  = (const v16bf*)Wsw;
    const v16bf* Bm = (const v16bf*)(nrm + (size_t)b * 8 * 256 * 512);
    v8f acc[2][4];
#pragma unroll
    for (int g = 0; g < 2; ++g)
#pragma unroll
        for (int t = 0; t < 4; ++t) acc[g][t] = (v8f){0.f,0.f,0.f,0.f,0.f,0.f,0.f,0.f};

    v16bf a0 = A[((op * 2 + 0) * 8 + 0) * 32 + lane];
    v16bf a1 = A[((op * 2 + 1) * 8 + 0) * 32 + lane];
    v16bf bb[4];
#pragma unroll
    for (int t = 0; t < 4; ++t) bb[t] = Bm[((size_t)0 * 256 + ng * 4 + t) * 32 + lane];

#pragma unroll
    for (int kc = 0; kc < 8; ++kc) {
        v16bf na0, na1, nb[4];
        if (kc < 7) {
            na0 = A[((op * 2 + 0) * 8 + kc + 1) * 32 + lane];
            na1 = A[((op * 2 + 1) * 8 + kc + 1) * 32 + lane];
#pragma unroll
            for (int t = 0; t < 4; ++t)
                nb[t] = Bm[((size_t)(kc + 1) * 256 + ng * 4 + t) * 32 + lane];
        }
#pragma unroll
        for (int t = 0; t < 4; ++t) {
            acc[0][t] = __builtin_amdgcn_wmma_f32_16x16x32_bf16(
                false, a0, false, bb[t], (short)0, acc[0][t], false, false);
            acc[1][t] = __builtin_amdgcn_wmma_f32_16x16x32_bf16(
                false, a1, false, bb[t], (short)0, acc[1][t], false, false);
        }
        if (kc < 7) {
            a0 = na0; a1 = na1;
#pragma unroll
            for (int t = 0; t < 4; ++t) bb[t] = nb[t];
        }
    }

    float* q = qkv + (size_t)b * 384 * NPIX;
    int col = lane & 15, hi = lane >> 4;
#pragma unroll
    for (int g = 0; g < 2; ++g) {
        int ot = op * 2 + g;
#pragma unroll
        for (int t = 0; t < 4; ++t) {
            int nbase = (ng * 4 + t) * 16 + col;
#pragma unroll
            for (int rr = 0; rr < 8; ++rr) {
                int o = ot * 16 + hi * 8 + rr;
                q[(size_t)o * NPIX + nbase] = acc[g][t][rr];
            }
        }
    }
}

// ---------------------------------------------------------------------------
// k softmax statistics over N per (b, inner-row): max and sum(exp).
// ---------------------------------------------------------------------------
__global__ void k_ksoftmax_stats(const float* __restrict__ qkv,
                                 float* __restrict__ kmax, float* __restrict__ ksum) {
    __shared__ float red[256];
    int row = blockIdx.x;                 // 0..2047
    int b = row >> 7, i = row & 127;
    const float* base = qkv + (size_t)b * 384 * NPIX + (size_t)(128 + i) * NPIX;
    int t = threadIdx.x;
    float m = -3.4e38f;
    for (int n = t; n < NPIX; n += 256) m = fmaxf(m, base[n]);
    red[t] = m; __syncthreads();
    for (int s = 128; s > 0; s >>= 1) { if (t < s) red[t] = fmaxf(red[t], red[t + s]); __syncthreads(); }
    float mx = red[0]; __syncthreads();
    float s = 0.f;
    for (int n = t; n < NPIX; n += 256) s += __expf(base[n] - mx);
    red[t] = s; __syncthreads();
    for (int ss = 128; ss > 0; ss >>= 1) { if (t < ss) red[t] += red[t + ss]; __syncthreads(); }
    if (t == 0) { kmax[row] = mx; ksum[row] = red[0]; }
}

// ---------------------------------------------------------------------------
// context[b,h,d,e] = sum_n softmax_k[n,d] * v[n,e] / N  via V_WMMA_F32_16X16X4_F32.
// A (16x4 f32): lane m=lane%16, two K values at k=(lane/16)*2 (+0,+1) -> contiguous n.
// B (4x16 f32): lane col e=lane%16, two K values likewise.
// ---------------------------------------------------------------------------
__global__ void k_context(const float* __restrict__ qkv, const float* __restrict__ kmax,
                          const float* __restrict__ ksum, float* __restrict__ ctx) {
    int wid  = (blockIdx.x * blockDim.x + threadIdx.x) >> 5;   // 0..255
    int lane = threadIdx.x & 31;
    int tile = wid & 3, bh = wid >> 2;
    int b = bh >> 2, h = bh & 3;
    int dt = tile >> 1, et = tile & 1;
    int d = dt * 16 + (lane & 15);
    int e = et * 16 + (lane & 15);
    int koff = (lane >> 4) * 2;
    const float* krow = qkv + (size_t)b * 384 * NPIX + (size_t)(128 + h * 32 + d) * NPIX;
    const float* vrow = qkv + (size_t)b * 384 * NPIX + (size_t)(256 + h * 32 + e) * NPIX;
    float mx  = kmax[b * 128 + h * 32 + d];
    float inv = 1.0f / ksum[b * 128 + h * 32 + d];
    v8f acc = (v8f){0.f,0.f,0.f,0.f,0.f,0.f,0.f,0.f};
    for (int nb = 0; nb < NPIX; nb += 4) {
        v2f ka = *(const v2f*)(krow + nb + koff);
        v2f vb = *(const v2f*)(vrow + nb + koff);
        v2f a;
        a[0] = __expf(ka[0] - mx) * inv;
        a[1] = __expf(ka[1] - mx) * inv;
        acc = __builtin_amdgcn_wmma_f32_16x16x4_f32(
            false, a, false, vb, (short)0, acc, false, false);
    }
    float* cb = ctx + ((size_t)b * 4 + h) * 1024;
    int col = lane & 15, hi = lane >> 4;
#pragma unroll
    for (int rr = 0; rr < 8; ++rr) {
        int dd = dt * 16 + hi * 8 + rr;
        cb[dd * 32 + et * 16 + col] = acc[rr] * (1.0f / NPIX);
    }
}

// ---------------------------------------------------------------------------
// apply: out[b,n,h,e] = sum_d ctx[d,e] * softmax_d(q)[n,d] / sqrt(32)
// ctx is staged to LDS with CDNA5 async global->LDS copies (ASYNCcnt path).
// Writes bf16 in B-fragment layout for GEMM2 (K = inner = h*32+e).
// ---------------------------------------------------------------------------
__global__ void k_apply(const float* __restrict__ qkv, const float* __restrict__ ctx,
                        unsigned short* __restrict__ attn) {
    __shared__ float c[32 * 32];
    int blk = blockIdx.x;                       // b*64 + h*16 + nchunk
    int nchunk = blk & 15, h = (blk >> 4) & 3, b = blk >> 6;
    int t = threadIdx.x;
    const float* cg = ctx + ((size_t)b * 4 + h) * 1024;
    // Async global -> LDS copy of the 4KB context tile, tracked by ASYNCcnt.
    // LDS dest address = low 32 bits of the generic shared pointer (LDS aperture
    // keeps the byte offset in addr[31:0]).
    for (int i = t; i < 1024; i += 256) {
        unsigned lds_off = (unsigned)(size_t)(&c[i]);
        const float* gp = cg + i;
        asm volatile("global_load_async_to_lds_b32 %0, %1, off"
                     :: "v"(lds_off), "v"(gp) : "memory");
    }
    asm volatile("s_wait_asynccnt 0x0" ::: "memory");
    __syncthreads();

    int n = nchunk * 256 + t;
    const float* qbase = qkv + (size_t)b * 384 * NPIX + (size_t)(h * 32) * NPIX + n;
    float qv[32];
    float mx = -3.4e38f;
#pragma unroll
    for (int d = 0; d < 32; ++d) { qv[d] = qbase[(size_t)d * NPIX]; mx = fmaxf(mx, qv[d]); }
    float s = 0.f;
#pragma unroll
    for (int d = 0; d < 32; ++d) { qv[d] = __expf(qv[d] - mx); s += qv[d]; }
    float sc = 0.17677669529663687f / s;        // (1/sqrt(32)) / sum
#pragma unroll
    for (int d = 0; d < 32; ++d) qv[d] *= sc;
    unsigned short* ab = attn + (size_t)b * 4 * 256 * 512;
    int ntile = n >> 4, lanecol = n & 15;
    for (int e = 0; e < 32; ++e) {
        float o = 0.f;
#pragma unroll
        for (int d = 0; d < 32; ++d) o += c[d * 32 + e] * qv[d];
        int lane = ((e >> 4) << 4) | lanecol;
        int j = e & 15;
        ab[(((size_t)h * 256 + ntile) * 512) + lane * 16 + j] = f2bf(o);
    }
}

// ---------------------------------------------------------------------------
// GEMM2 + bias + channel-LN + residual, fused.  One block = (b, n-tile of 16),
// 16 waves, each wave one 16x16 o-tile via bf16 WMMA; LN over 256 rows in LDS.
// All 8 fragments loaded up-front (single wait), then 4 wmmas.
// ---------------------------------------------------------------------------
__global__ void k_out(const unsigned short* __restrict__ WoutSw,
                      const unsigned short* __restrict__ attn,
                      const float* __restrict__ bout,
                      const float* __restrict__ x,
                      float* __restrict__ out) {
    __shared__ float tile[256 * 16];
    __shared__ float psum[32][16];
    __shared__ float psum2[32][16];
    __shared__ float mu_s[16], rs_s[16];
    int b = blockIdx.x >> 8;                    // grid = 16*256
    int ntile = blockIdx.x & 255;
    int t = threadIdx.x;                        // 512 threads = 16 waves
    int wv = t >> 5, lane = t & 31;
    const v16bf* A  = (const v16bf*)WoutSw;
    const v16bf* Bm = (const v16bf*)(attn + (size_t)b * 4 * 256 * 512);
    v16bf af[4], bf[4];
#pragma unroll
    for (int kc = 0; kc < 4; ++kc) {
        af[kc] = A[(wv * 4 + kc) * 32 + lane];
        bf[kc] = Bm[((size_t)kc * 256 + ntile) * 32 + lane];
    }
    v8f acc = (v8f){0.f,0.f,0.f,0.f,0.f,0.f,0.f,0.f};
#pragma unroll
    for (int kc = 0; kc < 4; ++kc) {
        acc = __builtin_amdgcn_wmma_f32_16x16x32_bf16(
            false, af[kc], false, bf[kc], (short)0, acc, false, false);
    }
    int col = lane & 15, hi = lane >> 4;
#pragma unroll
    for (int rr = 0; rr < 8; ++rr) {
        int o = wv * 16 + hi * 8 + rr;
        tile[o * 16 + col] = acc[rr] + bout[o];
    }
    __syncthreads();
    {   // partial sums: 32 parts x 16 cols, 8 rows each
        int cl = t & 15, part = t >> 4;
        float s = 0.f, s2 = 0.f;
#pragma unroll
        for (int rr = 0; rr < 8; ++rr) { float v = tile[(part * 8 + rr) * 16 + cl]; s += v; s2 += v * v; }
        psum[part][cl] = s; psum2[part][cl] = s2;
    }
    __syncthreads();
    if (t < 16) {
        float s = 0.f, s2 = 0.f;
        for (int p = 0; p < 32; ++p) { s += psum[p][t]; s2 += psum2[p][t]; }
        float mu  = s * (1.f / 256.f);
        float var = s2 * (1.f / 256.f) - mu * mu;
        mu_s[t] = mu; rs_s[t] = __frsqrt_rn(var + 1e-5f);
    }
    __syncthreads();
    {
        int cl = t & 15, part = t >> 4;
        int n = ntile * 16 + cl;
        float mu = mu_s[cl], rs = rs_s[cl];
#pragma unroll
        for (int rr = 0; rr < 8; ++rr) {
            int o = part * 8 + rr;
            float v = (tile[o * 16 + cl] - mu) * rs;
            size_t idx = (size_t)b * CH * NPIX + (size_t)o * NPIX + n;
            out[idx] = v + x[idx];
        }
    }
}

extern "C" void kernel_launch(void* const* d_in, const int* in_sizes, int n_in,
                              void* d_out, int out_size, void* d_ws, size_t ws_size,
                              hipStream_t stream) {
    const float* x    = (const float*)d_in[0];
    const float* Wqkv = (const float*)d_in[1];
    const float* Wout = (const float*)d_in[2];
    const float* bout = (const float*)d_in[3];
    float* out = (float*)d_out;
    char* ws = (char*)d_ws;
    unsigned short* nrm  = (unsigned short*)(ws + OFF_NORM);
    float*          qkv  = (float*)(ws + OFF_QKV);
    unsigned short* attn = (unsigned short*)(ws + OFF_ATTN);
    float*          ctx  = (float*)(ws + OFF_CTX);
    float*          kmax = (float*)(ws + OFF_KMAX);
    float*          ksum = (float*)(ws + OFF_KSUM);
    unsigned short* wqsw = (unsigned short*)(ws + OFF_WQKV);
    unsigned short* wosw = (unsigned short*)(ws + OFF_WOUT);

    k_wswz<<<(24 * 8 * 512 + 255) / 256, 256, 0, stream>>>(Wqkv, wqsw, 24, 8, 256);
    k_wswz<<<(16 * 4 * 512 + 255) / 256, 256, 0, stream>>>(Wout, wosw, 16, 4, 128);
    k_ln_swizzle<<<256, 256, 0, stream>>>(x, nrm);
    k_gemm_qkv<<<1536, 256, 0, stream>>>(wqsw, nrm, qkv);        // 12288 waves, 32x64/wave
    k_ksoftmax_stats<<<2048, 256, 0, stream>>>(qkv, kmax, ksum);
    k_context<<<128, 64, 0, stream>>>(qkv, kmax, ksum, ctx);     // 256 waves
    k_apply<<<1024, 256, 0, stream>>>(qkv, ctx, attn);
    k_out<<<4096, 512, 0, stream>>>(wosw, attn, bout, x, out);
}
